// CondensedLinearFineGrainedSparseOp_13597866459293
// MI455X (gfx1250) — compile-verified, matrix operand order checked by reading, not run on hardware
//
#include <hip/hip_runtime.h>

typedef __attribute__((ext_vector_type(16))) __bf16 v16bf;
typedef __attribute__((ext_vector_type(8)))  __bf16 v8bf;
typedef __attribute__((ext_vector_type(8)))  float  v8f;
typedef __attribute__((ext_vector_type(4)))  float  v4f;

namespace {
constexpr int kK = 4096;          // D_IN
constexpr int kN = 4096;          // D_OUT
constexpr int BM = 128;           // block tile M
constexpr int BN = 256;           // block tile N
constexpr int BK = 32;            // one wmma K-step per LDS stage
constexpr int LDS_STRIDE = 40;    // BK + 8 pad -> 80B rows: 16B aligned, bank-spread
constexpr int NKT = kK / BK;      // 128 stages
constexpr unsigned kStageElems = (unsigned)(BM + BN) * LDS_STRIDE;   // A then B
constexpr unsigned kStageBytes = kStageElems * 2;                    // 30720 B
constexpr int STAGES = 3;         // async ring depth (2 stages in flight)
}

// ---------------- pass 1: fp32 -> bf16 (one-shot, memory bound) ----------------
__global__ __launch_bounds__(256) void cvt_f32_to_bf16(
    const float* __restrict__ src, __bf16* __restrict__ dst, long long n) {
  long long i = ((long long)blockIdx.x * blockDim.x + threadIdx.x) * 8;
  if (i + 8 > n) return;
  v4f a = *(const v4f*)(src + i);
  v4f b = *(const v4f*)(src + i + 4);
  v8bf o;
  o[0] = (__bf16)a[0]; o[1] = (__bf16)a[1]; o[2] = (__bf16)a[2]; o[3] = (__bf16)a[3];
  o[4] = (__bf16)b[0]; o[5] = (__bf16)b[1]; o[6] = (__bf16)b[2]; o[7] = (__bf16)b[3];
  *(v8bf*)(dst + i) = o;
}

// CDNA5 async copy: 16B per lane, global -> LDS, tracked by ASYNCcnt.
// VDST = per-lane LDS byte address (HW adds wave LDS_BASE), VADDR = 64b global.
__device__ __forceinline__ void async_copy16(unsigned lds_addr, const __bf16* g) {
  asm volatile("global_load_async_to_lds_b128 %0, %1, off"
               :: "v"(lds_addr), "v"(g) : "memory");
}

// Async loads complete in order, so waiting asynccnt<=6 guarantees the oldest
// stage (6 copies) has landed while the newest stage stays in flight.
template <int N>
__device__ __forceinline__ void wait_async() {
#if __has_builtin(__builtin_amdgcn_s_wait_asynccnt)
  __builtin_amdgcn_s_wait_asynccnt(N);
#else
  asm volatile("s_wait_asynccnt %0" :: "i"(N) : "memory");
#endif
}

// One 16x16x32 bf16 fragment from an LDS row pointer (caller pre-offsets by
// halfsel*8 per the ISA A/B lane layout). Two ds_load_b128 + register concat.
__device__ __forceinline__ v16bf load_frag(const __bf16* p) {
  v8bf lo = *(const v8bf*)(p);
  v8bf hi = *(const v8bf*)(p + 16);
  return __builtin_shufflevector(lo, hi, 0, 1, 2, 3, 4, 5, 6, 7,
                                 8, 9, 10, 11, 12, 13, 14, 15);
}

// ---------------- pass 2: bf16 WMMA GEMM, fp32 accumulate ----------------
// Block: 256 threads = 8 waves, tile 128x256. Wave tile 64x64 = 4x4 fragments
// -> 16 wmma per 16 ds_load_b128 per stage. Global->LDS via async DMA in a
// 3-deep ring: issue stage kt+2, compute kt, wait asynccnt<=6 (kt+1 landed).
__global__ __launch_bounds__(256) void sparse_linear_wmma_bf16(
    const __bf16* __restrict__ X,   // [M, K] row-major (K contiguous)
    const __bf16* __restrict__ W,   // [N, K] row-major (K contiguous) == B^T
    const float* __restrict__ bias, // [N]
    float* __restrict__ out) {      // [M, N]
  __shared__ __bf16 lds[STAGES][kStageElems]; // per stage: A[128*40] then B[256*40]

  const int tid  = threadIdx.x;
  const int lane = tid & 31;
  const int l16  = lane & 15;
  const int hsel = lane >> 4;          // 0/1 half-wave select (K-chunk)
  const int wid  = tid >> 5;           // 8 waves
  const int wm   = (wid >> 2) * 64;    // 2 waves down
  const int wn   = (wid & 3) * 64;     // 4 waves across

  const long long mBlk = (long long)blockIdx.y * BM;
  const long long nBlk = (long long)blockIdx.x * BN;

  // Global->LDS staging map: thread owns 16B chunks at column gCol of
  // A rows {r, r+64} and B rows {r, r+64, r+128, r+192}, r = tid>>2.
  const int gRow = tid >> 2;           // 0..63
  const int gCol = (tid & 3) * 8;      // 0,8,16,24

  const __bf16* pa0 = X + (mBlk + gRow) * (long long)kK + gCol;
  const __bf16* pa1 = pa0 + 64LL * kK;
  const __bf16* pb0 = W + (nBlk + gRow) * (long long)kK + gCol;
  const __bf16* pb1 = pb0 + 64LL * kK;
  const __bf16* pb2 = pb0 + 128LL * kK;
  const __bf16* pb3 = pb0 + 192LL * kK;

  const unsigned ldsBase = (unsigned)(uintptr_t)&lds[0][0];
  const unsigned rowOff  = (unsigned)(gRow * LDS_STRIDE + gCol) * 2;
  const unsigned la0 = ldsBase + rowOff;
  const unsigned la1 = la0 + 64u * LDS_STRIDE * 2;
  const unsigned lb0 = ldsBase + (unsigned)BM * LDS_STRIDE * 2 + rowOff;
  const unsigned lb1 = lb0 + 64u * LDS_STRIDE * 2;
  const unsigned lb2 = lb0 + 128u * LDS_STRIDE * 2;
  const unsigned lb3 = lb0 + 192u * LDS_STRIDE * 2;

  v8f acc[4][4] = {};

  // Prologue: fill stages 0 and 1.
  async_copy16(la0, pa0);
  async_copy16(la1, pa1);
  async_copy16(lb0, pb0);
  async_copy16(lb1, pb1);
  async_copy16(lb2, pb2);
  async_copy16(lb3, pb3);
  pa0 += BK; pa1 += BK; pb0 += BK; pb1 += BK; pb2 += BK; pb3 += BK;
  async_copy16(la0 + kStageBytes, pa0);
  async_copy16(la1 + kStageBytes, pa1);
  async_copy16(lb0 + kStageBytes, pb0);
  async_copy16(lb1 + kStageBytes, pb1);
  async_copy16(lb2 + kStageBytes, pb2);
  async_copy16(lb3 + kStageBytes, pb3);
  wait_async<6>();     // stage 0 landed; stage 1 may still fly
  __syncthreads();

  unsigned bufE  = 0;                   // compute-stage offset (elements)
  unsigned fillB = 2u * kStageBytes;    // fill-stage offset (bytes)

  for (int kt = 0; kt < NKT; ++kt) {
    const bool issue2 = (kt + 2) < NKT;
    const bool need1  = (kt + 1) < NKT;

    // Kick off stage kt+2 into the ring slot last read at stage kt-1.
    if (issue2) {
      pa0 += BK; pa1 += BK; pb0 += BK; pb1 += BK; pb2 += BK; pb3 += BK;
      async_copy16(la0 + fillB, pa0);
      async_copy16(la1 + fillB, pa1);
      async_copy16(lb0 + fillB, pb0);
      async_copy16(lb1 + fillB, pb1);
      async_copy16(lb2 + fillB, pb2);
      async_copy16(lb3 + fillB, pb3);
    }

    const __bf16* pA = &lds[0][0] + bufE + hsel * 8;
    const __bf16* pB = pA + BM * LDS_STRIDE;
    v16bf af[4], bfr[4];
#pragma unroll
    for (int fm = 0; fm < 4; ++fm)
      af[fm] = load_frag(pA + (wm + fm * 16 + l16) * LDS_STRIDE);
#pragma unroll
    for (int fn = 0; fn < 4; ++fn)
      bfr[fn] = load_frag(pB + (wn + fn * 16 + l16) * LDS_STRIDE);

#pragma unroll
    for (int fm = 0; fm < 4; ++fm)
#pragma unroll
      for (int fn = 0; fn < 4; ++fn)
        acc[fm][fn] = __builtin_amdgcn_wmma_f32_16x16x32_bf16(
            false, af[fm], false, bfr[fn], (short)0, acc[fm][fn], false, false);

    // Stage kt+1 must have landed before anyone reads it after the barrier;
    // the just-issued kt+2 copies may remain outstanding (in-order completion).
    if (need1) {
      if (issue2) wait_async<6>(); else wait_async<0>();
    }
    __syncthreads();

    // Rotate ring offsets (uniform SALU).
    bufE  = (bufE  == 2u * kStageElems) ? 0u : bufE  + kStageElems;
    fillB = (fillB == 2u * kStageBytes) ? 0u : fillB + kStageBytes;
  }

  // Epilogue. C/D layout: lane l16 = N column; VGPR r = M row (r + hsel*8).
#pragma unroll
  for (int fm = 0; fm < 4; ++fm) {
    const long long mBase = mBlk + wm + fm * 16 + hsel * 8;
#pragma unroll
    for (int fn = 0; fn < 4; ++fn) {
      const long long n = nBlk + wn + fn * 16 + l16;
      const float bv = bias[n];
      float* po = out + mBase * kN + n;
#pragma unroll
      for (int r = 0; r < 8; ++r)
        po[(long long)r * kN] = acc[fm][fn][r] + bv;
    }
  }
}

// ---------------- fallback (if workspace too small or M % BM != 0) ----------------
__global__ __launch_bounds__(256) void naive_linear_f32(
    const float* __restrict__ X, const float* __restrict__ W,
    const float* __restrict__ bias, float* __restrict__ out, long long M) {
  long long idx = (long long)blockIdx.x * blockDim.x + threadIdx.x;
  if (idx >= M * kN) return;
  long long m = idx / kN, n = idx % kN;
  const float* xr = X + m * (long long)kK;
  const float* wr = W + n * (long long)kK;
  float s = 0.f;
  for (int k = 0; k < kK; k += 4) {
    v4f xv = *(const v4f*)(xr + k);
    v4f wv = *(const v4f*)(wr + k);
    s += xv[0] * wv[0] + xv[1] * wv[1] + xv[2] * wv[2] + xv[3] * wv[3];
  }
  out[idx] = s + bias[n];
}

extern "C" void kernel_launch(void* const* d_in, const int* in_sizes, int n_in,
                              void* d_out, int out_size, void* d_ws, size_t ws_size,
                              hipStream_t stream) {
  const float* x    = (const float*)d_in[0];   // [B,S,Din] fp32
  const float* w    = (const float*)d_in[1];   // [Dout,Din] fp32 (explicit zeros)
  const float* bias = (const float*)d_in[2];   // [Dout] fp32
  float* out = (float*)d_out;

  const long long Kd = kK;
  const long long M  = (long long)in_sizes[0] / Kd;   // B*S = 16384
  const long long nx = M * Kd;
  const long long nw = (long long)kN * Kd;
  const size_t need = (size_t)(nx + nw) * 2;          // bf16 workspace

  if (ws_size >= need && (M % BM) == 0) {
    __bf16* xb = (__bf16*)d_ws;
    __bf16* wb = (__bf16*)((char*)d_ws + (size_t)nx * 2);

    cvt_f32_to_bf16<<<dim3((unsigned)((nx / 8 + 255) / 256)), 256, 0, stream>>>(x, xb, nx);
    cvt_f32_to_bf16<<<dim3((unsigned)((nw / 8 + 255) / 256)), 256, 0, stream>>>(w, wb, nw);

    dim3 grid(kN / BN, (unsigned)(M / BM));
    sparse_linear_wmma_bf16<<<grid, 256, 0, stream>>>(xb, wb, bias, out);
  } else {
    long long total = M * kN;
    naive_linear_f32<<<dim3((unsigned)((total + 255) / 256)), 256, 0, stream>>>(
        x, w, bias, out, M);
  }
}
